// VectorQuantizer_70617852281236
// MI455X (gfx1250) — matrix operand order repaired and checked by reading, not compiled, not run on hardware
//
#include <hip/hip_runtime.h>

typedef __bf16 bf16_t;
typedef __bf16 v8bf  __attribute__((ext_vector_type(8)));
typedef __bf16 v16bf __attribute__((ext_vector_type(16)));
typedef float  v8f   __attribute__((ext_vector_type(8)));

#define D_DIM   1024
#define K_CODES 8192
#define N_ROWS  8192

#define BM 128
#define BN 128
#define BK 32
#define LSTR 40   // padded LDS row stride in bf16 elements (80 B, 16B-aligned, bank-conflict free)

// async copy: 16 bytes global -> LDS, tracked with ASYNCcnt (CDNA5 ISA 15.18.3)
#define ASYNC_LD16(ldsoff, gptr)                                             \
  asm volatile("global_load_async_to_lds_b128 %0, %1, off"                   \
               :: "v"(ldsoff), "v"(gptr) : "memory")

// ---------------------------------------------------------------------------
// prep: split fp32 rows into bf16 hi/lo parts + compute row squared norm
// ---------------------------------------------------------------------------
__global__ __launch_bounds__(256) void vq_prep(const float* __restrict__ src,
                                               bf16_t* __restrict__ hi,
                                               bf16_t* __restrict__ lo,
                                               float* __restrict__ norms) {
  const int row = blockIdx.x;
  const int tid = threadIdx.x;
  const size_t base = (size_t)row * D_DIM;
  float acc = 0.f;
  for (int d = tid; d < D_DIM; d += 256) {
    float x = src[base + d];
    bf16_t h = (bf16_t)x;
    bf16_t l = (bf16_t)(x - (float)h);
    hi[base + d] = h;
    lo[base + d] = l;
    acc += x * x;
  }
  __shared__ float red[256];
  red[tid] = acc;
  __syncthreads();
  for (int s = 128; s > 0; s >>= 1) {
    if (tid < s) red[tid] += red[tid + s];
    __syncthreads();
  }
  if (tid == 0) norms[row] = red[0];
}

// ---------------------------------------------------------------------------
// reset per-row packed (dist, col) argmin keys (ws is not re-poisoned between
// graph replays, so this must run every call)
// ---------------------------------------------------------------------------
__global__ __launch_bounds__(256) void vq_init_keys(unsigned long long* __restrict__ minkey) {
  const int i = blockIdx.x * 256 + threadIdx.x;
  if (i < N_ROWS) minkey[i] = ~0ull;
}

// ---------------------------------------------------------------------------
// helpers
// ---------------------------------------------------------------------------
static __device__ __forceinline__ v16bf make_v16(const bf16_t* p0, const bf16_t* p1) {
  v8bf a = *(const v8bf*)p0;
  v8bf b = *(const v8bf*)p1;
  v16bf r;
#pragma unroll
  for (int i = 0; i < 8; ++i) { r[i] = a[i]; r[8 + i] = b[i]; }
  return r;
}

static __device__ __forceinline__ unsigned long long shfl_xor_u64(unsigned long long v, int m) {
  unsigned lo = (unsigned)v, hi = (unsigned)(v >> 32);
  lo = (unsigned)__shfl_xor((int)lo, m, 32);
  hi = (unsigned)__shfl_xor((int)hi, m, 32);
  return ((unsigned long long)hi << 32) | lo;
}

// low 32 bits of the flat address of a __shared__ object == LDS byte offset
static __device__ __forceinline__ unsigned lds_off(const void* p) {
  return (unsigned)(size_t)p;
}

// ---------------------------------------------------------------------------
// GEMM: logits[n, k] = 2 * (z . e) - ||z||^2 - ||e||^2  via split-bf16 WMMA,
// async global->LDS staging, fused per-row argmin (packed u64 atomic min).
// block tile 128x128, 8 waves (2x4), wave tile 64x32 (4x2 WMMA accumulators)
// ---------------------------------------------------------------------------
__global__ __launch_bounds__(256) void vq_gemm(
    const bf16_t* __restrict__ zhi, const bf16_t* __restrict__ zlo,
    const bf16_t* __restrict__ ehi, const bf16_t* __restrict__ elo,
    const float*  __restrict__ zsq, const float*  __restrict__ esq,
    float* __restrict__ logits, unsigned long long* __restrict__ minkey) {
  __shared__ __align__(16) bf16_t sZh[BM * LSTR];
  __shared__ __align__(16) bf16_t sZl[BM * LSTR];
  __shared__ __align__(16) bf16_t sEh[BN * LSTR];
  __shared__ __align__(16) bf16_t sEl[BN * LSTR];

  const int tid     = threadIdx.x;
  const int lane    = tid & 31;
  const int wave    = tid >> 5;
  const int wm      = wave & 1;     // 0..1  -> 64-row slice
  const int wn      = wave >> 1;    // 0..3  -> 32-col slice
  const int l16     = lane & 15;
  const int halfsel = lane >> 4;    // 0 or 1
  const int a_koff  = halfsel * 8;  // A fragment: lanes>=16 start at K=8
  const int b_koff  = halfsel * 8;  // B fragment: lanes>=16 start at K=8

  const int m_blk = blockIdx.y * BM;
  const int n_blk = blockIdx.x * BN;

  // cooperative staging: each thread copies 16 bf16 per array per K-chunk
  const int lrow = tid >> 1;        // 0..127
  const int lcol = (tid & 1) * 16;  // 0 or 16
  const size_t gz = ((size_t)(m_blk + lrow) << 10) + lcol;
  const size_t ge = ((size_t)(n_blk + lrow) << 10) + lcol;
  // LDS destinations are loop-invariant: hoist their byte offsets
  const unsigned oZh = lds_off(&sZh[lrow * LSTR + lcol]);
  const unsigned oZl = lds_off(&sZl[lrow * LSTR + lcol]);
  const unsigned oEh = lds_off(&sEh[lrow * LSTR + lcol]);
  const unsigned oEl = lds_off(&sEl[lrow * LSTR + lcol]);

  v8f acc[4][2] = {};

  for (int k0 = 0; k0 < D_DIM; k0 += BK) {
    // ---- stage K-chunk into LDS via async copies (ASYNCcnt) ----------
    ASYNC_LD16(oZh,      zhi + gz + k0);
    ASYNC_LD16(oZh + 16, zhi + gz + k0 + 8);
    ASYNC_LD16(oZl,      zlo + gz + k0);
    ASYNC_LD16(oZl + 16, zlo + gz + k0 + 8);
    ASYNC_LD16(oEh,      ehi + ge + k0);
    ASYNC_LD16(oEh + 16, ehi + ge + k0 + 8);
    ASYNC_LD16(oEl,      elo + ge + k0);
    ASYNC_LD16(oEl + 16, elo + ge + k0 + 8);
    if (k0 + BK < D_DIM) {
      __builtin_prefetch(zhi + gz + k0 + BK, 0, 1);   // global_prefetch_b8
      __builtin_prefetch(ehi + ge + k0 + BK, 0, 1);
    }
    asm volatile("s_wait_asynccnt 0x0" ::: "memory");
    __syncthreads();

    // ---- load WMMA fragments per documented CDNA5 layouts ------------
    // A (16x32 bf16): lane L holds row M=L%16; elems 0..7 = K=a_koff+0..7,
    //                 elems 8..15 = K=a_koff+16..23
    v16bf ah[4], al[4];
#pragma unroll
    for (int i = 0; i < 4; ++i) {
      const bf16_t* pa = &sZh[(wm * 64 + i * 16 + l16) * LSTR + a_koff];
      const bf16_t* qa = &sZl[(wm * 64 + i * 16 + l16) * LSTR + a_koff];
      ah[i] = make_v16(pa, pa + 16);
      al[i] = make_v16(qa, qa + 16);
    }
    // B (32x16 bf16): lane L holds col N=L%16; elems 0..7 = K=b_koff+0..7,
    //                 elems 8..15 = K=b_koff+16..23
    v16bf bh[2], bl[2];
#pragma unroll
    for (int j = 0; j < 2; ++j) {
      const bf16_t* pb = &sEh[(wn * 32 + j * 16 + l16) * LSTR + b_koff];
      const bf16_t* qb = &sEl[(wn * 32 + j * 16 + l16) * LSTR + b_koff];
      bh[j] = make_v16(pb, pb + 16);
      bl[j] = make_v16(qb, qb + 16);
    }

    // ---- split-bf16 matmul: hi*hi + lo*hi + hi*lo (f32 accumulate) ---
#pragma unroll
    for (int i = 0; i < 4; ++i)
#pragma unroll
      for (int j = 0; j < 2; ++j) {
        acc[i][j] = __builtin_amdgcn_wmma_f32_16x16x32_bf16(
            false, ah[i], false, bh[j], (short)0, acc[i][j], false, false);
        acc[i][j] = __builtin_amdgcn_wmma_f32_16x16x32_bf16(
            false, al[i], false, bh[j], (short)0, acc[i][j], false, false);
        acc[i][j] = __builtin_amdgcn_wmma_f32_16x16x32_bf16(
            false, ah[i], false, bl[j], (short)0, acc[i][j], false, false);
      }
    __syncthreads();
  }

  // ---- epilogue: logits = 2*s - ||z||^2 - ||e||^2, fused argmin --------
  // C/D layout: VGPR r, lanes 0-15 -> M=r, N=lane; lanes 16-31 -> M=8+r
#pragma unroll
  for (int i = 0; i < 4; ++i) {
    const int rowb = m_blk + wm * 64 + i * 16 + halfsel * 8;
#pragma unroll
    for (int r = 0; r < 8; ++r) {
      const float zn = zsq[rowb + r];
      float bestd = 3.4e38f;
      int   bestc = 0;
#pragma unroll
      for (int j = 0; j < 2; ++j) {
        const int col = n_blk + wn * 32 + j * 16 + l16;
        const float logit = 2.f * acc[i][j][r] - zn - esq[col];
        logits[(size_t)(rowb + r) * K_CODES + col] = logit;
        const float dist = fmaxf(-logit, 0.f);   // >=0 -> bit-ordered
        if (dist < bestd) { bestd = dist; bestc = col; }  // j ascending -> first min
      }
      // pack (dist bits, col): u64 min == (min dist, lowest col on ties)
      unsigned long long key =
          ((unsigned long long)__float_as_uint(bestd) << 32) | (unsigned)bestc;
      // butterfly min across the 16 lanes sharing this row (xor<=8 never
      // crosses the lane-half boundary, so both halves reduce independently)
#pragma unroll
      for (int m = 1; m < 16; m <<= 1) {
        unsigned long long o = shfl_xor_u64(key, m);
        if (o < key) key = o;
      }
      if (l16 == 0) atomicMin(&minkey[rowb + r], key);
    }
  }
}

// ---------------------------------------------------------------------------
// finalize: decode argmin keys -> indices, gather quantized rows, loss = 0
// ---------------------------------------------------------------------------
__global__ __launch_bounds__(256) void vq_finalize(
    const unsigned long long* __restrict__ minkey,
    const float* __restrict__ codebook,
    float* __restrict__ quantized, int* __restrict__ indices,
    float* __restrict__ loss) {
  const int row = blockIdx.x;
  const int tid = threadIdx.x;
  const int idx = (int)(unsigned)(minkey[row] & 0xFFFFFFFFull);
  if (tid == 0) indices[row] = idx;
  const size_t cb = (size_t)idx * D_DIM;
  const size_t qb = (size_t)row * D_DIM;
  for (int d = tid; d < D_DIM; d += 256)
    quantized[qb + d] = codebook[cb + d];
  if (row == 0 && tid == 0) *loss = 0.f;
}

// ---------------------------------------------------------------------------
extern "C" void kernel_launch(void* const* d_in, const int* in_sizes, int n_in,
                              void* d_out, int out_size, void* d_ws, size_t ws_size,
                              hipStream_t stream) {
  const float* z  = (const float*)d_in[0];   // [8,1024,1024] f32
  const float* cb = (const float*)d_in[1];   // [8192,1024]  f32

  // workspace layout: 4 x 16MB bf16 arrays + 2 x 32KB norms + 64KB argmin keys
  bf16_t* zhi = (bf16_t*)d_ws;
  bf16_t* zlo = zhi + (size_t)N_ROWS * D_DIM;
  bf16_t* ehi = zlo + (size_t)N_ROWS * D_DIM;
  bf16_t* elo = ehi + (size_t)K_CODES * D_DIM;
  float*  zsq = (float*)(elo + (size_t)K_CODES * D_DIM);
  float*  esq = zsq + N_ROWS;
  unsigned long long* minkey = (unsigned long long*)(esq + K_CODES);

  // output layout (flat, in reference return order)
  float* out       = (float*)d_out;
  float* quantized = out;                                        // 8*1024*1024
  int*   indices   = (int*)(out + (size_t)N_ROWS * D_DIM);       // 8192 (i32 bits)
  float* loss      = out + (size_t)N_ROWS * D_DIM + N_ROWS;      // 1
  float* logits    = loss + 1;                                   // 8192*8192

  vq_init_keys<<<N_ROWS / 256, 256, 0, stream>>>(minkey);
  vq_prep<<<N_ROWS, 256, 0, stream>>>(z, zhi, zlo, zsq);
  vq_prep<<<K_CODES, 256, 0, stream>>>(cb, ehi, elo, esq);
  vq_gemm<<<dim3(K_CODES / BN, N_ROWS / BM), 256, 0, stream>>>(
      zhi, zlo, ehi, elo, zsq, esq, logits, minkey);
  vq_finalize<<<N_ROWS, 256, 0, stream>>>(minkey, cb, quantized, indices, loss);
}
